// TCMCProbability_5411658793262
// MI455X (gfx1250) — matrix operand order repaired and checked by reading, not compiled, not run on hardware
//
#include <hip/hip_runtime.h>

// ---------------------------------------------------------------------------
// Types
// ---------------------------------------------------------------------------
typedef __attribute__((ext_vector_type(16))) __bf16        v16bf;
typedef __attribute__((ext_vector_type(8)))  float         v8f;
typedef __attribute__((ext_vector_type(8)))  __bf16        bf16x8;
typedef __attribute__((ext_vector_type(2)))  float         v2f;
typedef __attribute__((ext_vector_type(4)))  unsigned int  u32x4;

union Frag16 {                 // one 16-bit WMMA operand (A or B), 8 VGPRs
    unsigned int u[8];
    u32x4        q[2];
    v16bf        v;
};

#if __has_builtin(__builtin_amdgcn_wmma_f32_16x16x4_f32)
#define HAVE_WMMA_F32X4 1
#else
#define HAVE_WMMA_F32X4 0
#endif

#define S 64
#define M 8
#define N_LEAVES 32
#define N_EDGES 62

// CDNA5 async global->LDS copy (ASYNCcnt-tracked, no VGPR round trip).
// VDST = VGPR holding LDS byte address, VADDR = 64-bit global address.
__device__ __forceinline__ void async_b128(unsigned lds_addr, const void* gsrc)
{
    asm volatile("global_load_async_to_lds_b128 %0, %1, off"
                 :: "v"(lds_addr), "v"((unsigned long long)gsrc)
                 : "memory");
}
__device__ __forceinline__ void wait_async0()
{
    asm volatile("s_wait_asynccnt 0x0" ::: "memory");
}

// ---------------------------------------------------------------------------
// Kernel 1: pi + generator Q  (one block per model)
// ---------------------------------------------------------------------------
__global__ __launch_bounds__(256) void setup_kernel(const float* __restrict__ R_inv,
                                                    const float* __restrict__ pi_inv,
                                                    float* __restrict__ Qg,
                                                    float* __restrict__ pi_ws)
{
    __shared__ float pis[S];
    __shared__ float Qs[S * S];
    const int m   = blockIdx.x;
    const int tid = threadIdx.x;

    if (tid == 0) {
        float n2 = 0.f;
        for (int j = 0; j < S - 1; ++j) { float p = pi_inv[m * (S - 1) + j]; n2 += p * p; }
        const float inv = 1.0f / (n2 + 1.0f);
        for (int j = 0; j < S - 1; ++j) {
            float p = 2.0f * pi_inv[m * (S - 1) + j] * inv;
            pis[j] = p * p;
        }
        float l = (n2 - 1.0f) * inv;
        pis[S - 1] = l * l;
    }
    __syncthreads();

    // off-diagonal Q_ij = exp(R[triu(i,j)]) * pi_j
    for (int idx = tid; idx < S * S; idx += 256) {
        const int i = idx >> 6, j = idx & 63;
        float q = 0.f;
        if (i != j) {
            const int i0 = i < j ? i : j;
            const int j0 = i < j ? j : i;
            const int ridx = 63 * i0 - (i0 * (i0 - 1)) / 2 + (j0 - i0 - 1);
            q = expf(R_inv[m * 2016 + ridx]) * pis[j];
        }
        Qs[idx] = q;
    }
    __syncthreads();
    if (tid < S) {               // diagonal = -row sum
        float rs = 0.f;
        for (int j = 0; j < S; ++j) rs += Qs[tid * S + j];
        Qs[tid * S + tid] = -rs;
    }
    __syncthreads();
    for (int idx = tid; idx < S * S; idx += 256) Qg[m * S * S + idx] = Qs[idx];
    if (tid < S) pi_ws[m * S + tid] = pis[tid];
}

// ---------------------------------------------------------------------------
// Kernel 1b: one-shot f32->bf16 leaf convert + transpose to [leaf][b][s].
// Removes the 8x redundant per-model conversion and makes each pruning
// block's leaf tile a set of contiguous 2KB chunks (async-copy friendly).
// ---------------------------------------------------------------------------
__global__ __launch_bounds__(256) void leafcvt_kernel(const float* __restrict__ leaves,
                                                      unsigned short* __restrict__ leavesT,
                                                      int B)
{
    const int idx = blockIdx.x * 256 + threadIdx.x;      // over B*32*64
    if (idx >= B * N_LEAVES * S) return;
    const int b    = idx >> 11;
    const int leaf = (idx >> 6) & 31;
    const int ss   = idx & 63;
    union { __bf16 h; unsigned short u; } cv;
    cv.h = (__bf16)leaves[idx];
    leavesT[((size_t)leaf * B + b) * S + ss] = cv.u;
}

// ---------------------------------------------------------------------------
// 64x64x64 f32 matmul in LDS.  WMMA f32 16x16x4 path (codegen-confirmed),
// VALU fallback otherwise.  Block = 128 threads (4 waves).
// ---------------------------------------------------------------------------
__device__ __forceinline__ void matmul64(float* __restrict__ C,
                                         const float* __restrict__ A,
                                         const float* __restrict__ B,
                                         int tid)
{
    __syncthreads();
#if HAVE_WMMA_F32X4
    const int lane = tid & 31;
    const int wave = tid >> 5;
    const int hi   = lane >> 4;            // lane group 0/1
    const int row  = wave * 16 + (lane & 15);
    const int nl   = lane & 15;
    for (int nt = 0; nt < 4; ++nt) {
        v8f acc = {};
        const int n = nt * 16 + nl;
        for (int kk = 0; kk < 16; ++kk) {
            const int kb = kk * 4 + hi * 2;            // K per A/B layout
            v2f a = *(const v2f*)&A[row * 64 + kb];    // A: V0=K(kb), V1=K(kb+1)
            v2f b;
            b.x = B[kb * 64 + n];
            b.y = B[(kb + 1) * 64 + n];
            acc = __builtin_amdgcn_wmma_f32_16x16x4_f32(false, a, false, b,
                                                        (short)0, acc, false, false);
        }
        const int cbase = wave * 16 + hi * 8;
        for (int r = 0; r < 8; ++r)
            C[(cbase + r) * 64 + n] = acc[r];
    }
#else
    const int row = tid >> 1;
    const int cb  = (tid & 1) * 32;
    float acc[32];
    for (int c = 0; c < 32; ++c) acc[c] = 0.f;
    for (int k = 0; k < 64; ++k) {
        const float a = A[row * 64 + k];
        for (int c = 0; c < 32; ++c) acc[c] = fmaf(a, B[k * 64 + cb + c], acc[c]);
    }
    for (int c = 0; c < 32; ++c) C[row * 64 + cb + c] = acc[c];
#endif
    __syncthreads();
}

// ---------------------------------------------------------------------------
// Kernel 2: P[e,m] = expm(t_e * Q_m), scaling & squaring + Taylor(10), f32.
// Result stored as bf16 row-major (feeds the WMMA pruning stage).
// Grid (62, 8), 128 threads.
// ---------------------------------------------------------------------------
__global__ __launch_bounds__(128) void expm_kernel(const float* __restrict__ Qg,
                                                   const float* __restrict__ lengths,
                                                   unsigned short* __restrict__ Pbf)
{
    __shared__ float As[4096], Pw[4096], Sm[4096], Tmp[4096];
    __shared__ float red[64];
    __shared__ int   s_sh;
    const int tid = threadIdx.x;
    const int e = blockIdx.x, m = blockIdx.y;
    const float t = lengths[e];

    for (int idx = tid; idx < 4096; idx += 128) As[idx] = Qg[m * 4096 + idx] * t;
    __syncthreads();

    if (tid < 64) {                      // inf-norm estimate
        float rs = 0.f;
        for (int j = 0; j < 64; ++j) rs += fabsf(As[tid * 64 + j]);
        red[tid] = rs;
    }
    __syncthreads();
    if (tid == 0) {
        float nrm = 0.f;
        for (int i = 0; i < 64; ++i) nrm = fmaxf(nrm, red[i]);
        int s = 0;
        while (nrm > 0.5f && s < 24) { nrm *= 0.5f; ++s; }
        s_sh = s;
    }
    __syncthreads();
    const int s = s_sh;
    const float sc = ldexpf(1.0f, -s);
    for (int idx = tid; idx < 4096; idx += 128) {
        const float a = As[idx] * sc;
        As[idx] = a;
        Pw[idx] = a;
        Sm[idx] = a + (((idx >> 6) == (idx & 63)) ? 1.0f : 0.0f);
    }
    __syncthreads();

    for (int k = 2; k <= 10; ++k) {      // Taylor series
        matmul64(Tmp, Pw, As, tid);
        const float inv = 1.0f / (float)k;
        for (int idx = tid; idx < 4096; idx += 128) {
            const float p = Tmp[idx] * inv;
            Pw[idx] = p;
            Sm[idx] += p;
        }
        __syncthreads();
    }
    for (int q = 0; q < s; ++q) {        // squaring
        matmul64(Tmp, Sm, Sm, tid);
        for (int idx = tid; idx < 4096; idx += 128) Sm[idx] = Tmp[idx];
        __syncthreads();
    }

    unsigned short* dst = Pbf + ((size_t)e * M + m) * 4096;
    for (int idx = tid; idx < 4096; idx += 128) {
        union { __bf16 h; unsigned short u; } cv;
        cv.h = (__bf16)Sm[idx];
        dst[idx] = cv.u;
    }
}

// ---------------------------------------------------------------------------
// Kernel 3: fused Felsenstein pruning.  One block = (model m, 16-col batch
// tile).  Leaf tile arrives via async global->LDS (bf16, pre-transposed);
// 5 level-synchronous rounds of bf16 WMMA (f32 accumulate) with sibling
// products in the accumulator.  Grid (B/16, 8), 128 threads (4 waves).
// ---------------------------------------------------------------------------
__global__ __launch_bounds__(128) void prune_kernel(const unsigned short* __restrict__ leavesT,
                                                    const float* __restrict__ pi_ws,
                                                    const unsigned short* __restrict__ Pb,
                                                    float* __restrict__ out,
                                                    int B)
{
    // bufA: 32 nodes x 16 batch x 64 states, bufB: 16 nodes  (bf16) = 96 KB
    __shared__ __bf16 sbuf[32 * 16 * 64 + 16 * 16 * 64];
    const int tid  = threadIdx.x;
    const int lane = tid & 31;
    const int wave = tid >> 5;
    const int nl   = lane & 15;
    const int hi   = lane >> 4;
    const int m    = blockIdx.y;
    const int bt   = blockIdx.x;

    // ---- async leaf tile load: per leaf one contiguous 2KB chunk ----------
    {
        const unsigned lds_base = (unsigned)(unsigned long long)(void*)sbuf;
        for (int leaf = wave; leaf < N_LEAVES; leaf += 4) {
            const char* src = (const char*)(leavesT + ((size_t)leaf * B + bt * 16) * S);
            const unsigned dst = lds_base + (unsigned)leaf * 2048u;
            for (int c = lane * 16; c < 2048; c += 32 * 16)
                async_b128(dst + (unsigned)c, src + c);
        }
        wait_async0();
    }
    __syncthreads();

    const int levelBase[5]  = {0, 32, 48, 56, 60};
    const int levelNodes[5] = {32, 16, 8, 4, 2};
    unsigned inOff = 0, outOff = 32 * 16 * 64;

    for (int L = 0; L < 5; ++L) {
        const int pairs = levelNodes[L] >> 1;
        for (int p = wave; p < pairs; p += 4) {
            const int c0 = 2 * p, c1 = 2 * p + 1;
            const __bf16* P0 = (const __bf16*)(Pb + ((size_t)(levelBase[L] + c0) * M + m) * 4096);
            const __bf16* P1 = (const __bf16*)(Pb + ((size_t)(levelBase[L] + c1) * M + m) * 4096);

            // prefetch this wave's next pair of P tiles (lane-spread over 8KB)
            if (p + 4 < pairs) {
                const char* n0 = (const char*)(Pb + ((size_t)(levelBase[L] + 2 * (p + 4)) * M + m) * 4096);
                const char* n1 = (const char*)(Pb + ((size_t)(levelBase[L] + 2 * (p + 4) + 1) * M + m) * 4096);
                __builtin_prefetch(n0 + lane * 256, 0, 0);
                __builtin_prefetch(n1 + lane * 256, 0, 0);
            }

            // B fragments (prev of each child), per K-half, reused across row tiles.
            // B in contiguous k order (same as A): lane n holds column n; lanes>=16 carry K+16;
            // per-VGPR packed K pairs are contiguous in row-major prev.
            Frag16 b0[2], b1[2];
            for (int kh = 0; kh < 2; ++kh) {
                const int kb = kh * 32 + hi * 16;
                const __bf16* r0 = &sbuf[inOff + (unsigned)(c0 * 16 + nl) * 64 + kb];
                const __bf16* r1 = &sbuf[inOff + (unsigned)(c1 * 16 + nl) * 64 + kb];
                b0[kh].q[0] = *(const u32x4*)(r0);
                b0[kh].q[1] = *(const u32x4*)(r0 + 8);
                b1[kh].q[0] = *(const u32x4*)(r1);
                b1[kh].q[1] = *(const u32x4*)(r1 + 8);
            }

            for (int rt = 0; rt < 4; ++rt) {
                v8f acc0 = {}, acc1 = {};
                const int row = rt * 16 + nl;
                for (int kh = 0; kh < 2; ++kh) {
                    // A in contiguous k order (matches B): V0..7 <- K 0..15
                    // (+16 for hi lanes); pairs contiguous.
                    const int koff = kh * 32 + hi * 16;
                    Frag16 a0, a1;
                    const __bf16* pa0 = P0 + row * 64 + koff;
                    const __bf16* pa1 = P1 + row * 64 + koff;
                    a0.q[0] = *(const u32x4*)(pa0);
                    a0.q[1] = *(const u32x4*)(pa0 + 8);
                    a1.q[0] = *(const u32x4*)(pa1);
                    a1.q[1] = *(const u32x4*)(pa1 + 8);
                    acc0 = __builtin_amdgcn_wmma_f32_16x16x32_bf16(false, a0.v, false, b0[kh].v,
                                                                   (short)0, acc0, false, false);
                    acc1 = __builtin_amdgcn_wmma_f32_16x16x32_bf16(false, a1.v, false, b1[kh].v,
                                                                   (short)0, acc1, false, false);
                }
                // sibling product in matching accumulator layout, repack bf16
                const v8f prod = acc0 * acc1;
                bf16x8 ob;
                for (int r = 0; r < 8; ++r) ob[r] = (__bf16)prod[r];
                const unsigned dst = outOff + (unsigned)(p * 16 + nl) * 64 + rt * 16 + hi * 8;
                *(bf16x8*)&sbuf[dst] = ob;
            }
        }
        __syncthreads();
        const unsigned t = inOff; inOff = outOff; outOff = t;
    }

    // root is at inOff (16 batch x 64 states); out[b,m] = <root[b,:], pi[m,:]>
    {
        float* red = (float*)sbuf;       // first 512 B of the other buffer
        const int i  = tid >> 3;
        const int s0 = (tid & 7) * 8;
        float partial = 0.f;
        for (int ss = 0; ss < 8; ++ss)
            partial += (float)sbuf[inOff + i * 64 + s0 + ss] * pi_ws[m * S + s0 + ss];
        red[tid] = partial;
        __syncthreads();
        if (tid < 16) {
            float sum = 0.f;
            for (int r = 0; r < 8; ++r) sum += red[tid * 8 + r];
            out[(size_t)(bt * 16 + tid) * M + m] = sum;
        }
    }
}

// ---------------------------------------------------------------------------
// Launch
// ---------------------------------------------------------------------------
extern "C" void kernel_launch(void* const* d_in, const int* in_sizes, int n_in,
                              void* d_out, int out_size, void* d_ws, size_t ws_size,
                              hipStream_t stream)
{
    const float* leaves  = (const float*)d_in[0];
    const float* R_inv   = (const float*)d_in[1];
    const float* pi_inv  = (const float*)d_in[2];
    const float* lengths = (const float*)d_in[3];
    (void)n_in; (void)out_size; (void)ws_size;

    const int B = in_sizes[0] / (N_LEAVES * S);

    char* ws = (char*)d_ws;
    unsigned short* Pbf     = (unsigned short*)ws;                   // 62*8*4096 bf16 (<4MB)
    unsigned short* leavesT = (unsigned short*)(ws + (4u << 20));    // B*32*64 bf16 (4MB @ B=1024)
    float* Qg  = (float*)(ws + (8u << 20));                          // 8*4096 f32
    float* piw = (float*)(ws + (8u << 20) + M * S * S * 4u);         // 8*64 f32
    float* out = (float*)d_out;

    setup_kernel<<<dim3(M), dim3(256), 0, stream>>>(R_inv, pi_inv, Qg, piw);
    leafcvt_kernel<<<dim3((B * N_LEAVES * S + 255) / 256), dim3(256), 0, stream>>>(leaves, leavesT, B);
    expm_kernel<<<dim3(N_EDGES, M), dim3(128), 0, stream>>>(Qg, lengths, Pbf);
    prune_kernel<<<dim3(B / 16, M), dim3(128), 0, stream>>>(leavesT, piw, Pbf, out, B);
}
